// GCN_81114752352945
// MI455X (gfx1250) — matrix operand order repaired and checked by reading, not compiled, not run on hardware
//
#include <hip/hip_runtime.h>
#include <hip/hip_bf16.h>

// ---------------------------------------------------------------------------
// GCN (2-layer) for MI455X / gfx1250.
//   layer: h = x @ W  (WMMA bf16, fp32 accum)
//          out[i] = b + dinv[i]^2 * h[i] + sum_{e: col=i} dinv[row]*w*dinv[col] * h[row]
//   relu between layers.
// ---------------------------------------------------------------------------

typedef __attribute__((ext_vector_type(16))) __bf16 v16bf;
typedef __attribute__((ext_vector_type(8)))  float  v8f;

struct Q2 { uint4 a, b; };   // 32 bytes, bit-cast container for a v16bf fragment

// ---- fp32 -> bf16 (round to nearest even), stored as u16 --------------------
__device__ __forceinline__ unsigned short f2bf(float f) {
    unsigned int u = __builtin_bit_cast(unsigned int, f);
    unsigned int r = (u + 0x7FFFu + ((u >> 16) & 1u)) >> 16;
    return (unsigned short)r;
}

// ---- degree / normalization -------------------------------------------------
__global__ void k_deg_init(float* deg, int n) {
    int i = blockIdx.x * blockDim.x + threadIdx.x;
    if (i < n) deg[i] = 1.0f;                       // self-loop weight
}

__global__ void k_deg_accum(const int* __restrict__ col, const float* __restrict__ w,
                            float* deg, int E) {
    int e = blockIdx.x * blockDim.x + threadIdx.x;
    if (e < E) atomicAdd(&deg[col[e]], w[e]);
}

__global__ void k_deg_finish(float* deg, int n) {
    int i = blockIdx.x * blockDim.x + threadIdx.x;
    if (i < n) {
        float d = deg[i];
        deg[i] = (d > 0.0f) ? rsqrtf(d) : 0.0f;     // in-place -> dinv
    }
}

// ---- fp32 -> bf16 cast of a flat array -------------------------------------
__global__ void k_cast_bf16(const float* __restrict__ src, unsigned short* __restrict__ dst,
                            int total) {
    int i = blockIdx.x * blockDim.x + threadIdx.x;
    if (i < total) dst[i] = f2bf(src[i]);
}

// ---- W[K][Ncols] fp32  ->  Wt[Ncols][K] bf16 --------------------------------
__global__ void k_transpose_w(const float* __restrict__ W, unsigned short* __restrict__ Wt,
                              int K, int Ncols) {
    int i = blockIdx.x * blockDim.x + threadIdx.x;
    if (i < K * Ncols) {
        int k = i / Ncols, nc = i % Ncols;
        Wt[nc * K + k] = f2bf(W[i]);
    }
}

// ---- WMMA GEMM: C[M][Ncols] = A[M][K](bf16) * Bt[Ncols][K](bf16)^T ----------
// One wave per 16x16 tile; K multiple of 32; Ncols multiple of 16.
__global__ void k_wmma_gemm(const unsigned short* __restrict__ A,
                            const unsigned short* __restrict__ Bt,
                            float* __restrict__ C,
                            int M, int K, int Ncols, int tilesN) {
    int wave  = (blockIdx.x * blockDim.x + threadIdx.x) >> 5;
    int lane  = threadIdx.x & 31;
    int tileM = wave / tilesN;
    int tileN = wave - tileM * tilesN;
    if (tileM * 16 >= M) return;                    // uniform per wave

    int half = lane >> 4;                           // 0 | 1
    int lr   = lane & 15;

    int rowA = tileM * 16 + lr; if (rowA >= M) rowA = M - 1;   // clamp (dup load)
    int rowB = tileN * 16 + lr;

    const uint4* ap = reinterpret_cast<const uint4*>(A + (size_t)rowA * K);
    const uint4* bp = reinterpret_cast<const uint4*>(Bt + (size_t)rowB * K);

    v8f acc = {};
    for (int kb = 0; kb < K; kb += 32) {
        // A fragment (16-bit A 16x32 layout): elems 0..7 = K[kb+8*half ..],
        //                                     elems 8..15 = K[kb+16+8*half ..]
        Q2 qa;
        qa.a = ap[(kb + half * 8) >> 3];
        qa.b = ap[(kb + 16 + half * 8) >> 3];
        // B fragment (same K layout as A): elems 0..7 = K[kb+8*half ..], elems 8..15 = K[kb+16+8*half ..]
        Q2 qb;
        qb.a = bp[(kb + half * 8) >> 3];
        qb.b = bp[(kb + 16 + half * 8) >> 3];

        v16bf av = __builtin_bit_cast(v16bf, qa);
        v16bf bv = __builtin_bit_cast(v16bf, qb);
        acc = __builtin_amdgcn_wmma_f32_16x16x32_bf16(
                  /*neg_a=*/false, av, /*neg_b=*/false, bv,
                  /*c_mod=*/(short)0, acc, /*reuse_a=*/false, /*reuse_b=*/false);
    }

    // C/D layout: VGPR r -> row (tileM*16 + r + 8*half), lane lr -> column
    int row0 = tileM * 16 + half * 8;
    int coln = tileN * 16 + lr;
    #pragma unroll
    for (int r = 0; r < 8; ++r) {
        int row = row0 + r;
        if (row < M) C[(size_t)row * Ncols + coln] = acc[r];
    }
}

// ---- out[i][f] = bias[f] + dinv[i]^2 * h[i][f]  (self-loop + bias init) ----
__global__ void k_agg_init(const float* __restrict__ h, const float* __restrict__ bias,
                           const float* __restrict__ dinv, float* __restrict__ out,
                           int n, int F) {
    int i = blockIdx.x * blockDim.x + threadIdx.x;
    if (i < n * F) {
        int node = i / F, f = i - node * F;
        float di = dinv[node];
        out[i] = bias[f] + di * di * h[i];
    }
}

// ---- edge scatter-add: one wave per edge, F/32 features per lane -----------
__global__ void k_agg_edges(const int* __restrict__ rowIdx, const int* __restrict__ colIdx,
                            const float* __restrict__ w, const float* __restrict__ dinv,
                            const float* __restrict__ h, float* __restrict__ out,
                            int E, int F) {
    int tid  = blockIdx.x * blockDim.x + threadIdx.x;
    int e    = tid >> 5;
    int lane = tid & 31;
    if (e >= E) return;
    int r = rowIdx[e], c = colIdx[e];
    float norm = dinv[r] * w[e] * dinv[c];
    int cpl = F >> 5;                                // features per lane
    const float* src = h   + (size_t)r * F + lane * cpl;
    float*       dst = out + (size_t)c * F + lane * cpl;
    #pragma unroll 4
    for (int j = 0; j < cpl; ++j)
        atomicAdd(&dst[j], norm * src[j]);
}

// ---- relu + cast to bf16 ----------------------------------------------------
__global__ void k_relu_cast(const float* __restrict__ src, unsigned short* __restrict__ dst,
                            int total) {
    int i = blockIdx.x * blockDim.x + threadIdx.x;
    if (i < total) {
        float v = src[i];
        dst[i] = f2bf(v > 0.0f ? v : 0.0f);
    }
}

// ---------------------------------------------------------------------------
static inline size_t alignUp(size_t x) { return (x + 255) & ~(size_t)255; }

extern "C" void kernel_launch(void* const* d_in, const int* in_sizes, int n_in,
                              void* d_out, int out_size, void* d_ws, size_t ws_size,
                              hipStream_t stream) {
    const float* x   = (const float*)d_in[0];
    const int*   ei  = (const int*)  d_in[1];   // [2][E]: rows then cols
    const float* ew  = (const float*)d_in[2];
    const float* W1  = (const float*)d_in[3];
    const float* b1  = (const float*)d_in[4];
    const float* W2  = (const float*)d_in[5];
    const float* b2  = (const float*)d_in[6];

    const int hid  = in_sizes[4];                 // 128
    const int in_c = in_sizes[3] / hid;           // 128
    const int outc = in_sizes[6];                 // 64
    const int n    = in_sizes[0] / in_c;          // 100000
    const int E    = in_sizes[2];                 // 1.6M

    const int* rowIdx = ei;
    const int* colIdx = ei + E;

    // ---- workspace carve ----
    char* w = (char*)d_ws;
    float* dinv = (float*)w;            w += alignUp((size_t)n * 4);
    const int maxF = (in_c > hid) ? in_c : hid;
    unsigned short* actBf = (unsigned short*)w;  w += alignUp((size_t)n * maxF * 2); // x_bf16, reused for relu(h1)
    float* h    = (float*)w;            w += alignUp((size_t)n * maxF * 4);          // GEMM out (h1 then h2)
    float* agg  = (float*)w;            w += alignUp((size_t)n * maxF * 4);          // layer-1 aggregation
    unsigned short* W1t = (unsigned short*)w;    w += alignUp((size_t)in_c * hid * 2);
    unsigned short* W2t = (unsigned short*)w;    w += alignUp((size_t)hid * outc * 2);
    (void)ws_size; (void)n_in; (void)out_size;

    const int B = 256;
    auto blk = [](long long t, int b) { return (unsigned)((t + b - 1) / b); };

    // ---- normalization coefficients ----
    k_deg_init  <<<blk(n, B), B, 0, stream>>>(dinv, n);
    k_deg_accum <<<blk(E, B), B, 0, stream>>>(colIdx, ew, dinv, E);
    k_deg_finish<<<blk(n, B), B, 0, stream>>>(dinv, n);

    // ---- weight prep (bf16, transposed to [N][K]) ----
    k_transpose_w<<<blk((long long)in_c * hid, B), B, 0, stream>>>(W1, W1t, in_c, hid);
    k_transpose_w<<<blk((long long)hid * outc, B), B, 0, stream>>>(W2, W2t, hid, outc);

    // ---- layer 1 ----
    k_cast_bf16<<<blk((long long)n * in_c, B), B, 0, stream>>>(x, actBf, n * in_c);
    {
        int tilesM = (n + 15) / 16, tilesN = hid / 16;
        long long threads = (long long)tilesM * tilesN * 32;
        k_wmma_gemm<<<blk(threads, B), B, 0, stream>>>(actBf, W1t, h, n, in_c, hid, tilesN);
    }
    k_agg_init <<<blk((long long)n * hid, B), B, 0, stream>>>(h, b1, dinv, agg, n, hid);
    k_agg_edges<<<blk((long long)E * 32, B), B, 0, stream>>>(rowIdx, colIdx, ew, dinv, h, agg, E, hid);
    k_relu_cast<<<blk((long long)n * hid, B), B, 0, stream>>>(agg, actBf, n * hid);

    // ---- layer 2 ----
    {
        int tilesM = (n + 15) / 16, tilesN = outc / 16;
        long long threads = (long long)tilesM * tilesN * 32;
        k_wmma_gemm<<<blk(threads, B), B, 0, stream>>>(actBf, W2t, h, n, hid, outc, tilesN);
    }
    float* out = (float*)d_out;
    k_agg_init <<<blk((long long)n * outc, B), B, 0, stream>>>(h, b2, dinv, out, n, outc);
    k_agg_edges<<<blk((long long)E * 32, B), B, 0, stream>>>(rowIdx, colIdx, ew, dinv, h, out, E, outc);
}
